// SE2AwareEdgeUpdateLayer_32212254720755
// MI455X (gfx1250) — compile-verified, matrix operand order checked by reading, not run on hardware
//
#include <hip/hip_runtime.h>
#include <hip/hip_bf16.h>

typedef __attribute__((ext_vector_type(16))) _Float16 v16h;
typedef __attribute__((ext_vector_type(8)))  _Float16 v8h;
typedef __attribute__((ext_vector_type(4)))  _Float16 h4;
typedef __attribute__((ext_vector_type(8)))  float    v8f;
typedef __attribute__((ext_vector_type(4)))  unsigned int u32x4;
typedef __attribute__((ext_vector_type(8)))  int      i32x8;
typedef __attribute__((ext_vector_type(4)))  int      i32x4;

#define H_DIM 128
#define TILE_E 64

#if __has_builtin(__builtin_amdgcn_tensor_load_to_lds)
#define HAVE_TDM 1
#else
#define HAVE_TDM 0
#endif

// ---- workspace layout (bytes) ----
#define WS_M10   0                     // 10*128 f32 = 5120
#define WS_FSEM  8192                  // 64 frags * 1KB = 65536
#define WS_FGATE (8192+65536)          // 32 frags = 32768
#define WS_FRES  (8192+65536+32768)    // 32 frags = 32768

// ---- LDS layout (bytes), dynamic ----
#define SM_A   0        // [64][264] f16 sem A panel; reused as hA [64][136] f16 = 33792
#define SM_F   33792    // [64][132] f32 geo -> h_pre -> h                       = 33792
#define SM_E   67584    // [64][136] f16 edge_attr f16 A panel                   = 17408
#define SM_RAW 84992    // [64][10]  f32                                         = 2560
#define SM_EA  87552    // [64][132] f32 edge_attr tile (TDM destination)        = 33792
#define SM_TOT 121344

// ---------------- prep: fold W_geo @ W_pair into M10 [10,128] ----------------
__global__ void prep_m10(const float* __restrict__ Wgeo, const float* __restrict__ Wpair,
                         float* __restrict__ M10) {
    int t = blockIdx.x * blockDim.x + threadIdx.x;
    if (t >= 10 * 128) return;
    int k = t >> 7, h = t & 127;
    int kk  = (k < 5) ? k : k - 5;
    int off = (k < 5) ? 0 : 128;
    float acc = 0.0f;
    for (int j = 0; j < 128; ++j)
        acc = fmaf(Wgeo[kk * 128 + j], Wpair[(off + j) * 128 + h], acc);
    M10[t] = acc;
}

// -------- prep: shuffle f32 weight [K,128] into per-lane WMMA B fragments ----
// frag(kt,nt): element(lane,h) = W[kt*32 + (h&7) + 16*(h>>3) + 8*(lane>>4)][nt*16 + (lane&15)]
__global__ void prep_frag(const float* __restrict__ W, _Float16* __restrict__ dst, int nfrag) {
    int t = blockIdx.x * blockDim.x + threadIdx.x;
    if (t >= nfrag * 512) return;
    int frag = t >> 9, w = t & 511;
    int lane = w >> 4, h = w & 15;
    int kt = frag >> 3, nt = frag & 7;
    int k   = kt * 32 + (h & 7) + ((h >> 3) << 4) + ((lane >> 4) << 3);
    int col = nt * 16 + (lane & 15);
    dst[t] = (_Float16)W[k * 128 + col];
}

// ------------------------------- main kernel --------------------------------
__global__ __launch_bounds__(128)
void se2_edge_kernel(const float* __restrict__ pos_i, const float* __restrict__ pos_j,
                     const float* __restrict__ hd_i,  const float* __restrict__ hd_j,
                     const int*   __restrict__ eidx,  const float* __restrict__ eattr,
                     const float* __restrict__ embi,  const float* __restrict__ embj,
                     const float* __restrict__ M10,
                     const _Float16* __restrict__ fsem,
                     const _Float16* __restrict__ fgate,
                     const _Float16* __restrict__ fres,
                     const float* __restrict__ bsem,
                     const float* __restrict__ gmm, const float* __restrict__ bta,
                     float* __restrict__ out, int E) {
    extern __shared__ unsigned char smem[];
    _Float16* bufA  = (_Float16*)(smem + SM_A);   // stride 264 halves
    float*    bufF  = (float*)   (smem + SM_F);   // stride 132 floats
    _Float16* bufE  = (_Float16*)(smem + SM_E);   // stride 136 halves
    float*    raw   = (float*)   (smem + SM_RAW); // stride 10 floats
    float*    bufEA = (float*)   (smem + SM_EA);  // stride 132 floats
    _Float16* hA    = bufA;                       // phase-3 overlay, stride 136 halves

    const int t  = threadIdx.x;
    const int e0 = blockIdx.x * TILE_E;
    int rows = E - e0; if (rows > TILE_E) rows = TILE_E;

#if HAVE_TDM
    // ---- Phase 0: TDM DMA of edge_attr tile [rows][128] f32 -> LDS (132-dword stride) ----
    if (t < 32) {
        unsigned long long ga = (unsigned long long)(const void*)(eattr + (size_t)e0 * H_DIM);
        u32x4 g0;
        g0.x = 1u;                                                   // count=1 (valid D#)
        g0.y = (unsigned)SM_EA;                                      // lds_addr (bytes)
        g0.z = (unsigned)(ga & 0xFFFFFFFFu);                         // global_addr[31:0]
        g0.w = (unsigned)((ga >> 32) & 0x01FFFFFFu) | (2u << 30);    // global_addr[56:32] | type=2
        i32x8 g1;
        g1[0] = (2 << 16)        // data_size = 4B
              | (1 << 20)        // pad_enable
              | (6 << 22)        // pad_interval: 128 dwords
              | (3 << 25);       // pad_amount:   4 dwords  -> 132-dword LDS row stride
        g1[1] = (H_DIM & 0xFFFF) << 16;                 // tensor_dim0[15:0]
        g1[2] = (H_DIM >> 16) | ((rows & 0xFFFF) << 16);// tensor_dim0[31:16] | tensor_dim1[15:0]
        g1[3] = (rows >> 16) | (H_DIM << 16);           // tensor_dim1[31:16] | tile_dim0
        g1[4] = TILE_E;                                 // tile_dim1 | tile_dim2=0
        g1[5] = H_DIM;                                  // tensor_dim0_stride[31:0]
        g1[6] = 0;
        g1[7] = 0;
        i32x4 gz4 = {0, 0, 0, 0};
        i32x8 gz8 = {0, 0, 0, 0, 0, 0, 0, 0};
        __builtin_amdgcn_tensor_load_to_lds(g0, g1, gz4, gz4, gz8, 0);
    }
#else
    { // fallback: stage edge_attr tile with vector loads
        int e = t >> 1, half = t & 1;
        int eg = e0 + e; if (eg >= E) eg = E - 1;
        const float4* rp = (const float4*)(eattr + (size_t)eg * H_DIM + half * 64);
        float4* dp = (float4*)(bufEA + e * 132 + half * 64);
        #pragma unroll 8
        for (int i = 0; i < 16; ++i) dp[i] = rp[i];
    }
#endif

    // ---- Phase 1a: geometry (10 scalars per edge) ----
    if (t < TILE_E) {
        int eg = e0 + t; if (eg >= E) eg = E - 1;
        int s = eidx[eg], g = eidx[E + eg];
        float pix = pos_i[2 * s], piy = pos_i[2 * s + 1];
        float pjx = pos_j[2 * g], pjy = pos_j[2 * g + 1];
        float hi = hd_i[s], hj = hd_j[g];
        float dx = pix - pjx, dy = piy - pjy;
        float dh = hi - hj;
        float cdh = __cosf(dh), sdh = __sinf(dh);
        float* rw = raw + t * 10;
        { // ij : frame of (pos_j[tgt], hd_j[tgt])
            float c = __cosf(hj), sn = __sinf(hj);
            float x = dx * c + dy * sn, y = -dx * sn + dy * c;
            rw[0] = sqrtf(x * x + y * y);
            float th = atan2f(y, x);
            rw[1] = __cosf(th); rw[2] = __sinf(th);
            rw[3] = cdh; rw[4] = sdh;
        }
        { // ji : d' = -d, frame of (pos_i[src], hd_i[src])
            float c = __cosf(hi), sn = __sinf(hi);
            float x = -dx * c - dy * sn, y = dx * sn - dy * c;
            rw[5] = sqrtf(x * x + y * y);
            float th = atan2f(y, x);
            rw[6] = __cosf(th); rw[7] = __sinf(th);
            rw[8] = cdh; rw[9] = -sdh;
        }
    }

    // ---- Phase 1b: gather+convert semantic A panel [64][256] f16 ----
    {
        int e = t >> 1, which = t & 1;
        int eg = e0 + e; if (eg >= E) eg = E - 1;
        int idx = which ? eidx[E + eg] : eidx[eg];
        const float4* rp = (const float4*)((which ? embj : embi) + (size_t)idx * H_DIM);
        h4* dp = (h4*)(bufA + e * 264 + which * 128);
        #pragma unroll 8
        for (int i = 0; i < 32; ++i) {
            float4 v = rp[i];
            h4 o = { (_Float16)v.x, (_Float16)v.y, (_Float16)v.z, (_Float16)v.w };
            dp[i] = o;
        }
    }

#if HAVE_TDM
    __builtin_amdgcn_s_wait_tensorcnt((short)0);   // DMA done (only wave 0 has TENSORcnt>0)
#endif
    __syncthreads();

    // ---- Phase 1c: edge_attr f32 (LDS) -> f16 A panel [64][128] ----
    {
        int e = t >> 1, half = t & 1;
        const float* rp = bufEA + e * 132 + half * 64;
        h4* dp = (h4*)(bufE + e * 136 + half * 64);
        #pragma unroll 4
        for (int i = 0; i < 16; ++i) {
            float4 v = *(const float4*)(rp + i * 4);
            h4 o = { (_Float16)v.x, (_Float16)v.y, (_Float16)v.z, (_Float16)v.w };
            dp[i] = o;
        }
    }

    // ---- Phase 1d: geo = relu(raw10 @ M10) -> bufF f32 ----
    {
        int e = t & 63, cb = (t >> 6) << 6;
        float r[10];
        #pragma unroll
        for (int k = 0; k < 10; ++k) r[k] = raw[e * 10 + k];
        for (int c = 0; c < 64; ++c) {
            int col = cb + c;
            float acc = 0.0f;
            #pragma unroll
            for (int k = 0; k < 10; ++k) acc = fmaf(r[k], M10[k * 128 + col], acc);
            bufF[e * 132 + col] = fmaxf(acc, 0.0f);
        }
    }
    __syncthreads();

    // ---- Phase 2: gate GEMM [64,256]x[256,128] on WMMA; fuse sigmoid*geo + edge_attr ----
    const int lane = t & 31, wv = t >> 5;
    const int m0 = wv * 16;
    const int cl = lane & 15;
    const int rsel = (lane >> 4) << 3; // 0 or 8

    v16h afr[8];                       // hoist A fragments: load LDS once, reuse for all nt
    #pragma unroll
    for (int kt = 0; kt < 8; ++kt) {
        const _Float16* ap = bufA + (m0 + cl) * 264 + kt * 32 + rsel;
        v8h a0 = *(const v8h*)ap;
        v8h a1 = *(const v8h*)(ap + 16);
        afr[kt] = __builtin_shufflevector(a0, a1, 0,1,2,3,4,5,6,7,8,9,10,11,12,13,14,15);
    }
    for (int nt = 0; nt < 8; ++nt) {
        v8f acc = {};
        #pragma unroll
        for (int kt = 0; kt < 8; ++kt) {
            v16h b = *(const v16h*)(fsem + (kt * 8 + nt) * 512 + lane * 16);
            acc = __builtin_amdgcn_wmma_f32_16x16x32_f16(false, afr[kt], false, b, (short)0, acc, false, false);
        }
        int col = nt * 16 + cl;
        float bias = bsem[col];
        #pragma unroll
        for (int i = 0; i < 8; ++i) {
            int row = m0 + i + rsel;
            float gate = 1.0f / (1.0f + __expf(-(acc[i] + bias)));
            float geo  = bufF[row * 132 + col];
            float ea   = bufEA[row * 132 + col];
            bufF[row * 132 + col] = geo * gate + ea;
        }
    }
    __syncthreads();

    // ---- Phase 2b: layernorm rows; write f32 back + f16 copy for W_gate GEMM ----
    if (t < TILE_E) {
        float* r = bufF + t * 132;
        float mu = 0.0f;
        for (int c = 0; c < 128; ++c) mu += r[c];
        mu *= (1.0f / 128.0f);
        float var = 0.0f;
        for (int c = 0; c < 128; ++c) { float d = r[c] - mu; var = fmaf(d, d, var); }
        float rs = rsqrtf(var * (1.0f / 128.0f) + 1e-5f);
        _Float16* hr = hA + t * 136;
        for (int c = 0; c < 128; ++c) {
            float hn = (r[c] - mu) * rs * gmm[c] + bta[c];
            r[c] = hn;
            hr[c] = (_Float16)hn;
        }
    }
    __syncthreads();

    // ---- Phase 3: update_gate = sig(h@W_gate), res = edge_attr@W_res; fuse output ----
    v16h ah[4], ae[4];                 // hoist both A panels into registers
    #pragma unroll
    for (int kt = 0; kt < 4; ++kt) {
        const _Float16* hp = hA   + (m0 + cl) * 136 + kt * 32 + rsel;
        const _Float16* ep = bufE + (m0 + cl) * 136 + kt * 32 + rsel;
        v8h h0 = *(const v8h*)hp, h1 = *(const v8h*)(hp + 16);
        v8h e0v = *(const v8h*)ep, e1v = *(const v8h*)(ep + 16);
        ah[kt] = __builtin_shufflevector(h0, h1, 0,1,2,3,4,5,6,7,8,9,10,11,12,13,14,15);
        ae[kt] = __builtin_shufflevector(e0v, e1v, 0,1,2,3,4,5,6,7,8,9,10,11,12,13,14,15);
    }
    for (int nt = 0; nt < 8; ++nt) {
        v8f accg = {}, accr = {};
        #pragma unroll
        for (int kt = 0; kt < 4; ++kt) {
            v16h bg = *(const v16h*)(fgate + (kt * 8 + nt) * 512 + lane * 16);
            v16h br = *(const v16h*)(fres  + (kt * 8 + nt) * 512 + lane * 16);
            accg = __builtin_amdgcn_wmma_f32_16x16x32_f16(false, ah[kt], false, bg, (short)0, accg, false, false);
            accr = __builtin_amdgcn_wmma_f32_16x16x32_f16(false, ae[kt], false, br, (short)0, accr, false, false);
        }
        int col = nt * 16 + cl;
        #pragma unroll
        for (int i = 0; i < 8; ++i) {
            int row = m0 + i + rsel;
            int ge = e0 + row;
            float h  = bufF[row * 132 + col];
            float ug = 1.0f / (1.0f + __expf(-accg[i]));
            float o  = h + ug * (accr[i] - h);
            if (ge < E) out[(size_t)ge * H_DIM + col] = o;
        }
    }
}

extern "C" void kernel_launch(void* const* d_in, const int* in_sizes, int n_in,
                              void* d_out, int out_size, void* d_ws, size_t ws_size,
                              hipStream_t stream) {
    const float* pos_i = (const float*)d_in[0];
    const float* pos_j = (const float*)d_in[1];
    const float* hd_i  = (const float*)d_in[2];
    const float* hd_j  = (const float*)d_in[3];
    const int*   eidx  = (const int*)  d_in[4];
    const float* eattr = (const float*)d_in[5];
    const float* embi  = (const float*)d_in[6];
    const float* embj  = (const float*)d_in[7];
    const float* Wgeo  = (const float*)d_in[8];
    const float* Wpair = (const float*)d_in[9];
    const float* Wsem  = (const float*)d_in[10];
    const float* bsem  = (const float*)d_in[11];
    const float* gmm   = (const float*)d_in[12];
    const float* bta   = (const float*)d_in[13];
    const float* Wgate = (const float*)d_in[14];
    const float* Wres  = (const float*)d_in[15];

    int E = in_sizes[4] / 2;

    char* ws = (char*)d_ws;
    float*    M10   = (float*)   (ws + WS_M10);
    _Float16* fsem  = (_Float16*)(ws + WS_FSEM);
    _Float16* fgate = (_Float16*)(ws + WS_FGATE);
    _Float16* fres  = (_Float16*)(ws + WS_FRES);

    prep_m10 <<<(10 * 128 + 255) / 256, 256, 0, stream>>>(Wgeo, Wpair, M10);
    prep_frag<<<(64 * 512) / 256, 256, 0, stream>>>(Wsem,  fsem,  64); // K=256
    prep_frag<<<(32 * 512) / 256, 256, 0, stream>>>(Wgate, fgate, 32); // K=128
    prep_frag<<<(32 * 512) / 256, 256, 0, stream>>>(Wres,  fres,  32); // K=128

    int blocks = (E + TILE_E - 1) / TILE_E;
    se2_edge_kernel<<<blocks, 128, SM_TOT, stream>>>(
        pos_i, pos_j, hd_i, hd_j, eidx, eattr, embi, embj,
        M10, fsem, fgate, fres, bsem, gmm, bta,
        (float*)d_out, E);
}
